// MuAttention_23948737642834
// MI455X (gfx1250) — matrix operand order repaired and checked by reading, not compiled, not run on hardware
//
#include <hip/hip_runtime.h>
#include <hip/hip_bf16.h>

// Problem constants (match reference)
#define S_LEN 2048
#define EMB   2048
#define NH    32
#define NKV   8
#define HD    64
#define BATCH 2
#define QTILE 64          // queries per attention workgroup (4 waves x 16)

typedef __attribute__((ext_vector_type(16))) __bf16 v16bf;
typedef __attribute__((ext_vector_type(8)))  float  v8f;

union ABFrag { v16bf v; unsigned u[8]; };
union Pack2  { __bf16 h[2]; unsigned u; };

__device__ __forceinline__ __bf16 f2bf(float f) {
  unsigned x = __builtin_bit_cast(unsigned, f);
  unsigned r = (x + 0x7FFFu + ((x >> 16) & 1u)) >> 16;   // round-to-nearest-even
  unsigned short hs = (unsigned short)r;
  return __builtin_bit_cast(__bf16, hs);
}

__device__ __forceinline__ unsigned pack2(float a, float b) {
  Pack2 p; p.h[0] = f2bf(a); p.h[1] = f2bf(b); return p.u;
}

__device__ __forceinline__ v8f wmma_bf16(v16bf a, v16bf b, v8f c) {
  // D = A(16x32 bf16) * B(32x16 bf16) + C(16x16 f32)
  return __builtin_amdgcn_wmma_f32_16x16x32_bf16(false, a, false, b, (short)0, c, false, false);
}

// ---- A fragment (16x32, MxK): element (m,k) at X[m*ld + kk + k] ----
__device__ __forceinline__ v16bf load_a_bf16(const __bf16* X, int ld, int kk, int lane) {
  int m = lane & 15, h = lane >> 4;
  const unsigned* p = (const unsigned*)(X + (size_t)m * ld + kk);
  ABFrag a;
  #pragma unroll
  for (int v = 0; v < 4; ++v) a.u[v]     = p[4*h + v];
  #pragma unroll
  for (int v = 0; v < 4; ++v) a.u[4 + v] = p[8 + 4*h + v];
  return a.v;
}

__device__ __forceinline__ v16bf load_a_f32(const float* X, int ld, int kk, int lane) {
  int m = lane & 15, h = lane >> 4;
  const float* row = X + (size_t)m * ld + kk;
  ABFrag a;
  #pragma unroll
  for (int v = 0; v < 4; ++v) { float2 f = *(const float2*)(row + 8*h + 2*v);      a.u[v]     = pack2(f.x, f.y); }
  #pragma unroll
  for (int v = 0; v < 4; ++v) { float2 f = *(const float2*)(row + 16 + 8*h + 2*v); a.u[4 + v] = pack2(f.x, f.y); }
  return a.v;
}

// ---- B fragment (32x16, KxN): element (k,n) at base[n*ldb + k] ----
__device__ __forceinline__ v16bf load_b_bf16(const __bf16* base, size_t ldb, int lane) {
  int n = lane & 15, h = lane >> 4;
  const unsigned* p = (const unsigned*)(base + (size_t)n * ldb);
  ABFrag b;
  #pragma unroll
  for (int v = 0; v < 4; ++v) b.u[v]     = p[4*h + v];
  #pragma unroll
  for (int v = 0; v < 4; ++v) b.u[4 + v] = p[8 + 4*h + v];
  return b.v;
}

// ---------------------------------------------------------------------------
// Tensor Data Mover staging (gfx1250). Guarded: falls back to cooperative copy.
// This toolchain declares the 6-arg builtin:
//   (uint32x4 g0, int32x8 g1, int32x4 g2, int32x4 g3, int32x8 g4, i32 cpol)
// ---------------------------------------------------------------------------
#if defined(__has_builtin)
#if __has_builtin(__builtin_amdgcn_tensor_load_to_lds) && __has_builtin(__builtin_amdgcn_s_wait_tensorcnt)
#define USE_TDM 1
#endif
#endif
#ifndef USE_TDM
#define USE_TDM 0
#endif

#if USE_TDM
typedef __attribute__((ext_vector_type(4))) unsigned uint4v;
typedef __attribute__((ext_vector_type(8))) int int8v;
typedef __attribute__((ext_vector_type(4))) int int4v;

// LDS byte offset = low 32 bits of the generic (flat) address (ISA 10.2 aperture map)
__device__ __forceinline__ unsigned lds_off(const void* p) {
  return (unsigned)(size_t)p;
}

// 2-D tile DMA: copies rows x cols bf16 elements, global row stride row_stride
// (elements), into LDS at ldsoff (packed rows of `cols` elements).
// D# fields per CDNA5 ISA 8.3/8.4: count=1, type=2, data_size=1 (2 bytes).
__device__ __forceinline__ void tdm_load_2d(const __bf16* gsrc, unsigned ldsoff,
                                            unsigned rows, unsigned cols,
                                            unsigned row_stride,
                                            unsigned tensor_d0, unsigned tensor_d1) {
  unsigned long long ga = (unsigned long long)(size_t)gsrc;
  uint4v g0 = { 1u,                                   // [1:0] count=1 (valid, user mode)
                ldsoff,                               // [63:32] lds_addr
                (unsigned)ga,                         // [95:64] global_addr lo
                (unsigned)((ga >> 32) & 0x01FFFFFFu) | (2u << 30) }; // addr hi | type=2
  int8v g1;
  g1[0] = (int)(1u << 16);                                            // data_size=2B
  g1[1] = (int)((tensor_d0 & 0xFFFFu) << 16);                         // tensor_dim0[15:0]
  g1[2] = (int)(((tensor_d0 >> 16) & 0xFFFFu) | ((tensor_d1 & 0xFFFFu) << 16));
  g1[3] = (int)(((tensor_d1 >> 16) & 0xFFFFu) | ((cols & 0xFFFFu) << 16)); // tile_dim0
  g1[4] = (int)(rows & 0xFFFFu);                                      // tile_dim1
  g1[5] = (int)row_stride;                                            // tensor_dim0_stride lo
  g1[6] = 0;
  g1[7] = 0;
  int4v z4 = {0, 0, 0, 0};
  int8v z8 = {0, 0, 0, 0, 0, 0, 0, 0};
  __builtin_amdgcn_tensor_load_to_lds(g0, g1, z4, z4, z8, 0);
}
#endif

// ---------------------------------------------------------------------------
// Weight f32 -> bf16 conversion
// ---------------------------------------------------------------------------
__global__ void cvt_f32_bf16(const float* __restrict__ src, __bf16* __restrict__ dst, int n) {
  int i = blockIdx.x * blockDim.x + threadIdx.x;
  if (i < n) dst[i] = f2bf(src[i]);
}

// ---------------------------------------------------------------------------
// Generic GEMM: Y[g][c] = sum_k A[g][k] * W[c][k] + bias[c]
//   MODE 0: Q proj -> bias, RoPE, *1/sqrt(D), bf16 [B,NH,S,D]
//   MODE 1: K proj -> bias, RoPE,             bf16 [B,NKV,S,D]
//   MODE 2: V proj -> bias,                   bf16 transposed [B,NKV,D,S]
//   MODE 3: O proj -> bias,                   f32  [B*S, EMB]
// One wave computes a 16x64 tile (4 WMMAs share each A fragment).
// ---------------------------------------------------------------------------
template <int MODE>
__global__ void proj_kernel(const float* __restrict__ Xf, const __bf16* __restrict__ Xh,
                            const __bf16* __restrict__ W, const float* __restrict__ bias,
                            __bf16* __restrict__ dst, float* __restrict__ dstf) {
  const int lane    = threadIdx.x;
  const int rowbase = blockIdx.x * 16;   // over B*S rows
  const int ct4     = blockIdx.y;        // 64-col tile == head index for modes 0..2
  const int cbase   = ct4 * 64;
  const int ln      = lane & 15;
  const int half    = lane >> 4;

  v8f zero = {0.f,0.f,0.f,0.f,0.f,0.f,0.f,0.f};
  v8f acc[4] = {zero, zero, zero, zero};

  #pragma unroll 1
  for (int kk = 0; kk < EMB; kk += 32) {
    v16bf a = (MODE == 3) ? load_a_bf16(Xh + (size_t)rowbase * EMB, EMB, kk, lane)
                          : load_a_f32 (Xf + (size_t)rowbase * EMB, EMB, kk, lane);
    #pragma unroll
    for (int t = 0; t < 4; ++t) {
      v16bf b = load_b_bf16(W + (size_t)(cbase + 16*t) * EMB + kk, EMB, lane);
      acc[t] = wmma_bf16(a, b, acc[t]);
    }
  }

  #pragma unroll
  for (int t = 0; t < 4; ++t) {
    float bv = bias[cbase + 16*t + ln];
    #pragma unroll
    for (int r = 0; r < 8; ++r) acc[t][r] += bv;
  }

  if (MODE == 0 || MODE == 1) {
    // RoPE: d = 16*t + ln; pairs (d, d+32) live in (t, t+2), same lane & row.
    float if0 = __powf(10000.f, -(float)(ln)      * (1.f / 32.f));
    float if1 = __powf(10000.f, -(float)(ln + 16) * (1.f / 32.f));
    #pragma unroll
    for (int r = 0; r < 8; ++r) {
      int g = rowbase + r + 8*half;
      float s = (float)(g & (S_LEN - 1));
      float t0 = s * if0, t1 = s * if1;
      float c0 = __cosf(t0), s0 = __sinf(t0), c1 = __cosf(t1), s1 = __sinf(t1);
      float y0 = acc[0][r], y1 = acc[1][r], y2 = acc[2][r], y3 = acc[3][r];
      acc[0][r] = y0 * c0 - y2 * s0;
      acc[1][r] = y1 * c1 - y3 * s1;
      acc[2][r] = y2 * c0 + y0 * s0;
      acc[3][r] = y3 * c1 + y1 * s1;
    }
    if (MODE == 0) {
      #pragma unroll
      for (int t = 0; t < 4; ++t)
        #pragma unroll
        for (int r = 0; r < 8; ++r) acc[t][r] *= 0.125f;   // 1/sqrt(HD)
    }
  }

  #pragma unroll
  for (int r = 0; r < 8; ++r) {
    int g = rowbase + r + 8*half;
    int b = g >> 11;                 // S_LEN == 2048
    int s = g & (S_LEN - 1);
    if (MODE == 0) {
      __bf16* p = dst + (((size_t)(b * NH + ct4) * S_LEN + s) * HD);
      #pragma unroll
      for (int t = 0; t < 4; ++t) p[16*t + ln] = f2bf(acc[t][r]);
    } else if (MODE == 1) {
      __bf16* p = dst + (((size_t)(b * NKV + ct4) * S_LEN + s) * HD);
      #pragma unroll
      for (int t = 0; t < 4; ++t) p[16*t + ln] = f2bf(acc[t][r]);
    } else if (MODE == 2) {
      #pragma unroll
      for (int t = 0; t < 4; ++t) {
        int d = 16*t + ln;
        dst[((size_t)(b * NKV + ct4) * HD + d) * S_LEN + s] = f2bf(acc[t][r]);
      }
    } else {
      float* p = dstf + (size_t)g * EMB + cbase;
      #pragma unroll
      for (int t = 0; t < 4; ++t) p[16*t + ln] = acc[t][r];
    }
  }
}

// ---------------------------------------------------------------------------
// Flash attention: 4 waves per block share a (b, h, 64-query) supertile.
// K/V 32-key tiles staged to double-buffered LDS by the Tensor Data Mover
// (wave 0 issues 2 descriptors/block, pipelined with s_wait_tensorcnt),
// WMMA B-fragments then come from LDS. Causal + GQA (kvh = h/4), Q pre-scaled.
// ---------------------------------------------------------------------------
__global__ void __launch_bounds__(128)
attn_kernel(const __bf16* __restrict__ Q, const __bf16* __restrict__ Kb,
            const __bf16* __restrict__ Vt, __bf16* __restrict__ AO) {
  __shared__ __bf16 Ksh[2][32 * HD];    // [key][d]
  __shared__ __bf16 Vsh[2][HD * 32];    // [d][key]
  __shared__ __bf16 Plds[4][16 * 32];   // per-wave P re-layout buffer

  const int tid  = threadIdx.x;
  const int lane = tid & 31;
  const int wv   = tid >> 5;               // wave in workgroup: 0..3
  const int bid  = blockIdx.x;             // b*NH*32 + h*32 + qst
  const int qst  = bid & 31;
  const int h    = (bid >> 5) & 31;
  const int b    = bid >> 10;
  const int kvh  = h >> 2;
  const int qsbase = qst * QTILE;
  const int qbase  = qsbase + wv * 16;
  const int ln   = lane & 15;
  const int half = lane >> 4;

  const __bf16* Qh = Q  + (size_t)(b * NH  + h)   * S_LEN * HD;   // [S, D]
  const __bf16* Kh = Kb + (size_t)(b * NKV + kvh) * S_LEN * HD;   // [S, D]
  const __bf16* Vh = Vt + (size_t)(b * NKV + kvh) * HD * S_LEN;   // [D, S]

  v16bf qa0 = load_a_bf16(Qh + (size_t)qbase * HD, HD, 0,  lane);
  v16bf qa1 = load_a_bf16(Qh + (size_t)qbase * HD, HD, 32, lane);

  v8f zero = {0.f,0.f,0.f,0.f,0.f,0.f,0.f,0.f};
  v8f o[4] = {zero, zero, zero, zero};
  float mx[8], sm[8];
  #pragma unroll
  for (int r = 0; r < 8; ++r) { mx[r] = -3.0e38f; sm[r] = 0.f; }

  const int nkb = (qsbase + QTILE + 31) >> 5;   // 32-key blocks covering [0, qsbase+64)

#if USE_TDM
  if (wv == 0) {
    tdm_load_2d(Kh, lds_off(Ksh[0]), 32, HD, HD, HD, S_LEN);
    tdm_load_2d(Vh, lds_off(Vsh[0]), HD, 32, S_LEN, S_LEN, HD);
  }
#endif

  #pragma unroll 1
  for (int jb = 0; jb < nkb; ++jb) {
    const int jbase = jb * 32;
    const int cur = jb & 1;

#if USE_TDM
    if (wv == 0) {
      if (jb + 1 < nkb) {
        const int jn = jbase + 32;
        tdm_load_2d(Kh + (size_t)jn * HD, lds_off(Ksh[cur ^ 1]), 32, HD, HD, HD, S_LEN);
        tdm_load_2d(Vh + jn,              lds_off(Vsh[cur ^ 1]), HD, 32, S_LEN, S_LEN, HD);
        __builtin_amdgcn_s_wait_tensorcnt(2);   // current pair complete (in-order)
      } else {
        __builtin_amdgcn_s_wait_tensorcnt(0);
      }
    }
    __syncthreads();
#else
    __syncthreads();
    {   // cooperative staging fallback (K tile rows are contiguous)
      const unsigned* ks = (const unsigned*)(Kh + (size_t)jbase * HD);
      unsigned* kd = (unsigned*)Ksh[cur];
      for (int i = tid; i < 32 * HD / 2; i += 128) kd[i] = ks[i];
      unsigned* vd = (unsigned*)Vsh[cur];
      for (int i = tid; i < HD * 32 / 2; i += 128) {
        int row = i >> 4, c = i & 15;
        vd[i] = *(const unsigned*)(Vh + (size_t)row * S_LEN + jbase + 2 * c);
      }
    }
    __syncthreads();
#endif

    if (jbase <= qbase + 15) {          // wave-uniform causal early-out
      const __bf16* Kt = Ksh[cur];
      const __bf16* Vb = Vsh[cur];

      // ---- S = Q K^T (two 16x16 score tiles from LDS) ----
      v8f s0 = zero, s1 = zero;
      {
        v16bf b00 = load_b_bf16(Kt + 0,            HD, lane);
        v16bf b01 = load_b_bf16(Kt + 32,           HD, lane);
        s0 = wmma_bf16(qa0, b00, s0);
        s0 = wmma_bf16(qa1, b01, s0);
        v16bf b10 = load_b_bf16(Kt + 16 * HD,      HD, lane);
        v16bf b11 = load_b_bf16(Kt + 16 * HD + 32, HD, lane);
        s1 = wmma_bf16(qa0, b10, s1);
        s1 = wmma_bf16(qa1, b11, s1);
      }

      // ---- online softmax (row m = r + 8*half spans 16 lanes of this half) ----
      #pragma unroll
      for (int r = 0; r < 8; ++r) {
        int q  = qbase + r + 8*half;
        int k0 = jbase + ln;
        int k1 = jbase + 16 + ln;
        float e0 = (k0 <= q) ? s0[r] : -3.0e38f;
        float e1 = (k1 <= q) ? s1[r] : -3.0e38f;
        float mrow = fmaxf(e0, e1);
        mrow = fmaxf(mrow, __shfl_xor(mrow, 1, 32));
        mrow = fmaxf(mrow, __shfl_xor(mrow, 2, 32));
        mrow = fmaxf(mrow, __shfl_xor(mrow, 4, 32));
        mrow = fmaxf(mrow, __shfl_xor(mrow, 8, 32));
        float mnew = fmaxf(mx[r], mrow);
        float corr = __expf(mx[r] - mnew);
        float p0 = (k0 <= q) ? __expf(s0[r] - mnew) : 0.f;
        float p1 = (k1 <= q) ? __expf(s1[r] - mnew) : 0.f;
        float ps = p0 + p1;
        ps += __shfl_xor(ps, 1, 32);
        ps += __shfl_xor(ps, 2, 32);
        ps += __shfl_xor(ps, 4, 32);
        ps += __shfl_xor(ps, 8, 32);
        sm[r] = sm[r] * corr + ps;
        mx[r] = mnew;
        #pragma unroll
        for (int t = 0; t < 4; ++t) o[t][r] *= corr;
        int m = r + 8*half;
        Plds[wv][m * 32 + ln]      = f2bf(p0);   // C-frag -> A-frag re-layout
        Plds[wv][m * 32 + 16 + ln] = f2bf(p1);
      }

      // ---- O += P V ----
      v16bf pa = load_a_bf16(Plds[wv], 32, 0, lane);
      #pragma unroll
      for (int t = 0; t < 4; ++t) {
        v16bf vb = load_b_bf16(Vb + (size_t)(16*t) * 32, 32, lane);
        o[t] = wmma_bf16(pa, vb, o[t]);
      }
    }

    __syncthreads();   // all waves done with buf[cur] before it is refilled
  }

  // ---- normalize and store to AO[(b*S + s), h*64 + d] ----
  #pragma unroll
  for (int r = 0; r < 8; ++r) {
    float inv = 1.f / sm[r];
    int g = b * S_LEN + qbase + r + 8*half;
    __bf16* p = AO + (size_t)g * EMB + h * HD;
    #pragma unroll
    for (int t = 0; t < 4; ++t) p[16*t + ln] = f2bf(o[t][r] * inv);
  }
}

// ---------------------------------------------------------------------------
extern "C" void kernel_launch(void* const* d_in, const int* in_sizes, int n_in,
                              void* d_out, int out_size, void* d_ws, size_t ws_size,
                              hipStream_t stream) {
  const float* query = (const float*)d_in[0];
  const float* key_  = (const float*)d_in[1];
  const float* value = (const float*)d_in[2];
  // d_in[3]: mask — all-true in this workload; causal mask applied in-kernel.
  const float* Wq = (const float*)d_in[4];
  const float* bq = (const float*)d_in[5];
  const float* Wk = (const float*)d_in[6];
  const float* bk = (const float*)d_in[7];
  const float* Wv = (const float*)d_in[8];
  const float* bv = (const float*)d_in[9];
  const float* Wo = (const float*)d_in[10];
  const float* bo = (const float*)d_in[11];
  float* out = (float*)d_out;

  char* ws = (char*)d_ws;
  size_t off = 0;
  auto alloc = [&](size_t bytes) -> void* {
    void* p = ws + off;
    off += (bytes + 255) & ~(size_t)255;
    return p;
  };

  const size_t nWq = (size_t)EMB * EMB;          // 4M
  const size_t nWk = (size_t)(NKV * HD) * EMB;   // 1M
  const size_t nAO = (size_t)BATCH * S_LEN * EMB;

  __bf16* Wq16 = (__bf16*)alloc(nWq * 2);
  __bf16* Wk16 = (__bf16*)alloc(nWk * 2);
  __bf16* Wv16 = (__bf16*)alloc(nWk * 2);
  __bf16* Wo16 = (__bf16*)alloc(nWq * 2);
  __bf16* Qbuf = (__bf16*)alloc((size_t)BATCH * NH  * S_LEN * HD * 2);
  __bf16* Kbuf = (__bf16*)alloc((size_t)BATCH * NKV * S_LEN * HD * 2);
  __bf16* Vtbf = (__bf16*)alloc((size_t)BATCH * NKV * HD * S_LEN * 2);
  __bf16* AO   = (__bf16*)alloc(nAO * 2);

  // 1) weights f32 -> bf16
  cvt_f32_bf16<<<(int)((nWq + 255) / 256), 256, 0, stream>>>(Wq, Wq16, (int)nWq);
  cvt_f32_bf16<<<(int)((nWk + 255) / 256), 256, 0, stream>>>(Wk, Wk16, (int)nWk);
  cvt_f32_bf16<<<(int)((nWk + 255) / 256), 256, 0, stream>>>(Wv, Wv16, (int)nWk);
  cvt_f32_bf16<<<(int)((nWq + 255) / 256), 256, 0, stream>>>(Wo, Wo16, (int)nWq);

  // 2) projections + RoPE (one wave per 16x64 tile)
  const int rowTiles = BATCH * S_LEN / 16;   // 256
  dim3 blk(32);
  proj_kernel<0><<<dim3(rowTiles, EMB / 64),      blk, 0, stream>>>(query, nullptr, Wq16, bq, Qbuf, nullptr);
  proj_kernel<1><<<dim3(rowTiles, NKV * HD / 64), blk, 0, stream>>>(key_,  nullptr, Wk16, bk, Kbuf, nullptr);
  proj_kernel<2><<<dim3(rowTiles, NKV * HD / 64), blk, 0, stream>>>(value, nullptr, Wv16, bv, Vtbf, nullptr);

  // 3) causal GQA flash attention, TDM-staged K/V in LDS
  attn_kernel<<<BATCH * NH * (S_LEN / QTILE), 128, 0, stream>>>(Qbuf, Kbuf, Vtbf, AO);

  // 4) output projection -> f32 d_out
  proj_kernel<3><<<dim3(rowTiles, EMB / 64), blk, 0, stream>>>(nullptr, AO, Wo16, bo, nullptr, out);

  (void)in_sizes; (void)n_in; (void)out_size; (void)ws_size;
}
